// StandardAttention_24833500905643
// MI455X (gfx1250) — matrix operand order repaired and checked by reading, not compiled, not run on hardware
//
#include <hip/hip_runtime.h>
#include <stdint.h>
#include <stddef.h>

// ---------------------------------------------------------------------------
// Types / helpers
// ---------------------------------------------------------------------------
typedef __attribute__((ext_vector_type(8)))  float        v8f;
typedef __attribute__((ext_vector_type(16))) __bf16       v16bf;
typedef __attribute__((ext_vector_type(4)))  unsigned int u32x4;
typedef __attribute__((ext_vector_type(8)))  unsigned int u32x8;
typedef __attribute__((ext_vector_type(4)))  int          i32x4;
typedef __attribute__((ext_vector_type(8)))  int          i32x8;

#if defined(__HIP_DEVICE_COMPILE__) && __has_builtin(__builtin_amdgcn_tensor_load_to_lds)
#define HAVE_TDM 1
#else
#define HAVE_TDM 0
#endif

__device__ __forceinline__ unsigned short f2bf(float f) {
  unsigned int u = __float_as_uint(f);
  return (unsigned short)((u + 0x7FFFu + ((u >> 16) & 1u)) >> 16);
}

__device__ __forceinline__ v8f vzero8() {
  v8f r;
  for (int i = 0; i < 8; ++i) r[i] = 0.0f;
  return r;
}

__device__ __forceinline__ v16bf frag16(u32x4 lo, u32x4 hi) {
  u32x8 u;
  u[0] = lo[0]; u[1] = lo[1]; u[2] = lo[2]; u[3] = lo[3];
  u[4] = hi[0]; u[5] = hi[1]; u[6] = hi[2]; u[7] = hi[3];
  return __builtin_bit_cast(v16bf, u);
}

__device__ __forceinline__ v16bf ldfrag32(const unsigned short* p) {
  u32x8 u = *(const u32x8*)p;
  return __builtin_bit_cast(v16bf, u);
}

__device__ __forceinline__ v8f wmma_bf16(v16bf a, v16bf b, v8f c) {
  // D = A(16x32 bf16) * B(32x16 bf16) + C(16x16 f32)
  return __builtin_amdgcn_wmma_f32_16x16x32_bf16(false, a, false, b,
                                                 (short)0, c, false, false);
}

// ---------------------------------------------------------------------------
// Weight packing: [Cout][Cin][kh][kw] f32 -> B-fragment-major bf16
//   index = ((kb*NcCols + n)*2 + h)*16 + e , k = kb*32 + h*16 + e
//   For a fixed (kb, n-tile) the 512 elements are CONTIGUOUS (1KB) -> one TDM
//   2D-tile row per K-step, row stride NcCols*64 bytes.
// ---------------------------------------------------------------------------
__global__ void k_pack_w(const float* __restrict__ w, unsigned short* __restrict__ wpk,
                         int Cin, int Cout, int ks, int NcCols, long total) {
  long idx = (long)blockIdx.x * 256 + threadIdx.x;
  if (idx >= total) return;
  int e = (int)(idx & 15);
  long t = idx >> 4;
  int h = (int)(t & 1); t >>= 1;
  int n  = (int)(t % NcCols);
  int kb = (int)(t / NcCols);
  int k = kb * 32 + h * 16 + e;
  float val = 0.0f;
  if (n < Cout) {
    int tap = k / Cin, ci = k % Cin;
    int ky = tap / ks, kx = tap % ks;
    val = w[(((size_t)n * Cin + ci) * ks + ky) * ks + kx];
  }
  wpk[idx] = f2bf(val);
}

// ---------------------------------------------------------------------------
// TDM stage: copy `rows` weight-fragment rows (1KB each, global row stride
// `strideBytes`) into LDS at byte offset ldsOff.  2D tensor descriptor,
// data_size = 4B, tile_dim0 = 256 dwords.
// ---------------------------------------------------------------------------
#if HAVE_TDM
__device__ __forceinline__ void tdm_stage_weights(const unsigned short* gsrc,
                                                  unsigned ldsOff, int rows,
                                                  int strideBytes) {
  unsigned long long ga = (unsigned long long)(size_t)gsrc;
  int strideDw = strideBytes >> 2;
  u32x4 g0;
  g0[0] = 1u;                                            // count=1, user D#
  g0[1] = ldsOff;                                        // lds_addr (bytes)
  g0[2] = (unsigned)ga;                                  // global_addr lo
  g0[3] = (unsigned)((ga >> 32) & 0x01FFFFFFu) | (2u << 30);  // addr hi | type=2
  i32x8 g1;
  g1[0] = (2 << 16);                                     // data_size = 4B
  g1[1] = (strideDw & 0xFFFF) << 16;                     // tensor_dim0 lo16
  g1[2] = ((strideDw >> 16) & 0xFFFF) | ((rows & 0xFFFF) << 16); // dim0 hi | dim1 lo
  g1[3] = (256 << 16);                                   // dim1 hi=0 | tile_dim0=256
  g1[4] = rows & 0xFFFF;                                 // tile_dim1=rows, tile_dim2=0
  g1[5] = strideDw;                                      // tensor_dim0_stride lo32
  g1[6] = 0;                                             // stride hi | dim1_stride lo
  g1[7] = 0;
  i32x4 g2; g2[0] = 0; g2[1] = 0; g2[2] = 0; g2[3] = 0;
  i32x4 g3 = g2;
  for (int i = 0; i < 4; ++i)
    g0[i] = (unsigned)__builtin_amdgcn_readfirstlane((int)g0[i]);
  for (int i = 0; i < 8; ++i)
    g1[i] = __builtin_amdgcn_readfirstlane(g1[i]);
#if __clang_major__ >= 23
  i32x8 g4; for (int i = 0; i < 8; ++i) g4[i] = 0;
  __builtin_amdgcn_tensor_load_to_lds(g0, g1, g2, g3, g4, 0);
#else
  __builtin_amdgcn_tensor_load_to_lds(g0, g1, g2, g3, 0);
#endif
}
#endif

// ---------------------------------------------------------------------------
// Generic implicit-GEMM conv (3x3 pad1 / 1x1 pad0), NHWC bf16 activations.
// Block = 8 waves sharing one N-tile (same weights); wave w owns M-tile
// (blockIdx/Ntiles)*8+w.  Weights staged to LDS via Tensor Data Mover in
// 32-K-row (32KB) stages; B fragments are then 32B ds loads.
// ---------------------------------------------------------------------------
struct ConvCfg {
  int M, H, W, HW;
  int hwShift, wShift, cinShift;
  int Cin, Cout, K, pad;
  int Ntiles, act, storeNCHW;
};

__device__ __forceinline__ u32x4 conv_lda(const unsigned short* __restrict__ act,
                                          int b, int y, int x, int k,
                                          const ConvCfg& c, int cmask) {
  int tap = k >> c.cinShift;
  int ci  = k & cmask;
  int ky = tap / 3;
  int kx = tap - ky * 3;
  int iy = y + ky - c.pad, ix = x + kx - c.pad;
  u32x4 z; z[0] = z[1] = z[2] = z[3] = 0u;
  if ((unsigned)iy >= (unsigned)c.H || (unsigned)ix >= (unsigned)c.W) return z;
  const unsigned short* p =
      act + (size_t)((b << c.hwShift) + (iy << c.wShift) + ix) * c.Cin + ci;
  return *(const u32x4*)p;
}

#define STAGE_ROWS 32   // K-rows per TDM stage (32KB of LDS)

__global__ __launch_bounds__(256)
void k_conv_wmma(const unsigned short* __restrict__ act,
                 const unsigned short* __restrict__ wpk,
                 const float* __restrict__ bias,
                 const float* __restrict__ bnS,
                 const float* __restrict__ bnB,
                 const float* __restrict__ slope,
                 float* __restrict__ outF,
                 unsigned short* __restrict__ outBf,
                 ConvCfg cfg) {
  __shared__ __align__(16) unsigned short wsm[STAGE_ROWS * 512];
  const int lane = threadIdx.x & 31, wave = threadIdx.x >> 5;
  const int hh = lane >> 4, ln = lane & 15;
  const int nt = blockIdx.x % cfg.Ntiles;
  const int mt = (blockIdx.x / cfg.Ntiles) * 8 + wave;
  const int m = mt * 16 + ln;
  const int b = m >> cfg.hwShift;
  const int pix = m & (cfg.HW - 1);
  const int y = pix >> cfg.wShift;
  const int x = pix & (cfg.W - 1);
  const int cmask = cfg.Cin - 1;
  const int NcCols = cfg.Ntiles << 4;
  const int strideBytes = NcCols * 64;          // global bytes between K-rows
  const unsigned short* gsrc0 = wpk + (size_t)nt * 512;   // this N-tile's chunk

  v8f acc0 = vzero8(), acc1 = vzero8();
  const int totRows = cfg.K >> 5;
  for (int r0 = 0; r0 < totRows; r0 += STAGE_ROWS) {
    const int rows = min(STAGE_ROWS, totRows - r0);       // always even here
    const unsigned short* gsrc = gsrc0 + (size_t)r0 * NcCols * 32;
    if (wave == 0) {
#if HAVE_TDM
      tdm_stage_weights(gsrc, (unsigned)(size_t)wsm, rows, strideBytes);
      __builtin_amdgcn_s_wait_tensorcnt(0);
#else
      // cooperative fallback copy (one wave): rows*256 dwords
      const unsigned int* src32 = (const unsigned int*)gsrc;
      unsigned int* dst32 = (unsigned int*)wsm;
      for (int t = lane; t < rows * 256; t += 32) {
        int rr = t >> 8, dd = t & 255;
        dst32[rr * 256 + dd] = src32[(size_t)rr * (strideBytes >> 2) + dd];
      }
#endif
    }
    __syncthreads();
    for (int r = 0; r < rows; r += 2) {
      const int kb0 = (r0 + r) * 32;
      u32x4 a0 = conv_lda(act, b, y, x, kb0 + hh * 16, cfg, cmask);
      u32x4 a1 = conv_lda(act, b, y, x, kb0 + hh * 16 + 8, cfg, cmask);
      v16bf af0 = frag16(a0, a1);
      v16bf bf0 = ldfrag32(wsm + (size_t)r * 512 + ln * 32 + hh * 16);
      acc0 = wmma_bf16(af0, bf0, acc0);
      const int kb1 = kb0 + 32;
      u32x4 a2 = conv_lda(act, b, y, x, kb1 + hh * 16, cfg, cmask);
      u32x4 a3 = conv_lda(act, b, y, x, kb1 + hh * 16 + 8, cfg, cmask);
      v16bf af1 = frag16(a2, a3);
      v16bf bf1 = ldfrag32(wsm + (size_t)(r + 1) * 512 + ln * 32 + hh * 16);
      acc1 = wmma_bf16(af1, bf1, acc1);
    }
    __syncthreads();
  }
  v8f acc = acc0 + acc1;

  const int n = nt * 16 + ln;
  const bool nvalid = n < cfg.Cout;
  const int nc = nvalid ? n : 0;
  float scale = 1.0f, shift = 0.0f;
  if (bias) shift = bias[nc];
  if (bnS) { float s = bnS[nc]; shift = shift * s + bnB[nc]; scale = s; }
  float a = (cfg.act == 2 && slope) ? slope[0] : 0.0f;
  for (int v = 0; v < 8; ++v) {
    int row = mt * 16 + v + 8 * hh;
    float val = acc[v] * scale + shift;
    if (cfg.act == 1) val = fmaxf(val, 0.0f);
    else if (cfg.act == 2) val = val > 0.0f ? val : a * val;
    if (nvalid) {
      if (outF) {
        if (cfg.storeNCHW) {
          int bb = row >> cfg.hwShift, pp = row & (cfg.HW - 1);
          outF[((size_t)bb * cfg.Cout + n) * cfg.HW + pp] = val;
        } else {
          outF[(size_t)row * cfg.Cout + n] = val;
        }
      }
      if (outBf) outBf[(size_t)row * cfg.Cout + n] = f2bf(val);
    }
  }
}

// ---------------------------------------------------------------------------
// Elementwise kernels
// ---------------------------------------------------------------------------
__global__ void k_concat(const float* __restrict__ x, const float* __restrict__ fu,
                         unsigned short* __restrict__ out, long total) {
  long idx = (long)blockIdx.x * 256 + threadIdx.x;
  if (idx >= total) return;
  int c = (int)(idx & 127);
  long p = idx >> 7;              // b*4096 + pix
  int b = (int)(p >> 12);
  int pix = (int)(p & 4095);
  const float* src = (c < 64) ? x : fu;
  int cc = c & 63;
  out[idx] = f2bf(src[((size_t)b * 64 + cc) * 4096 + pix]);
}

__global__ void k_maxpool2(const float* __restrict__ in, float* __restrict__ outF,
                           unsigned short* __restrict__ outBf,
                           int Hi, int Wi, int C, long total) {
  long idx = (long)blockIdx.x * 256 + threadIdx.x;
  if (idx >= total) return;
  int Wo = Wi >> 1;
  int c = (int)(idx % C);
  long t = idx / C;
  int ox = (int)(t % Wo); t /= Wo;
  int oy = (int)(t % (Hi >> 1));
  int b  = (int)(t / (Hi >> 1));
  size_t base = (((size_t)b * Hi + 2 * oy) * Wi + 2 * ox) * C + c;
  float m0 = fmaxf(in[base], in[base + C]);
  float m1 = fmaxf(in[base + (size_t)Wi * C], in[base + (size_t)Wi * C + C]);
  float m = fmaxf(m0, m1);
  if (outF)  outF[idx]  = m;
  if (outBf) outBf[idx] = f2bf(m);
}

__global__ void k_upsample2(const float* __restrict__ in, unsigned short* __restrict__ outBf,
                            int Hi, int Wi, int C, long total) {
  long idx = (long)blockIdx.x * 256 + threadIdx.x;
  if (idx >= total) return;
  int Ho = 2 * Hi, Wo = 2 * Wi;
  int c = (int)(idx % C);
  long t = idx / C;
  int ox = (int)(t % Wo); t /= Wo;
  int oy = (int)(t % Ho);
  int b  = (int)(t / Ho);
  float sy = (oy + 0.5f) * 0.5f - 0.5f;
  float sx = (ox + 0.5f) * 0.5f - 0.5f;
  int y0 = (int)floorf(sy), x0 = (int)floorf(sx);
  float wy = sy - (float)y0, wx = sx - (float)x0;
  int y1 = min(Hi - 1, max(0, y0 + 1)), x1 = min(Wi - 1, max(0, x0 + 1));
  y0 = min(Hi - 1, max(0, y0));  x0 = min(Wi - 1, max(0, x0));
  const float* base = in + (size_t)b * Hi * Wi * C + c;
  float v00 = base[((size_t)y0 * Wi + x0) * C];
  float v01 = base[((size_t)y0 * Wi + x1) * C];
  float v10 = base[((size_t)y1 * Wi + x0) * C];
  float v11 = base[((size_t)y1 * Wi + x1) * C];
  float v = (1.f - wy) * ((1.f - wx) * v00 + wx * v01) +
            wy * ((1.f - wx) * v10 + wx * v11);
  outBf[idx] = f2bf(v);
}

__global__ void k_pad_qk(const float* __restrict__ q, unsigned short* __restrict__ Qp,
                         long total) {
  long idx = (long)blockIdx.x * 256 + threadIdx.x;
  if (idx >= total) return;
  int c = (int)(idx & 31);
  long r = idx >> 5;
  Qp[idx] = (c < 8) ? f2bf(q[r * 8 + c]) : (unsigned short)0;
}

// NHWC f32 [B,4096,64] -> channel-major bf16 [B][64][4096]
__global__ void k_transpose_cn(const float* __restrict__ in, unsigned short* __restrict__ out,
                               long total) {
  long idx = (long)blockIdx.x * 256 + threadIdx.x;
  if (idx >= total) return;
  int i = (int)(idx % 4096);
  long t = idx / 4096;
  int c = (int)(t % 64);
  int b = (int)(t / 64);
  out[idx] = f2bf(in[((size_t)b * 4096 + i) * 64 + c]);
}

__global__ void k_fuse(const float* __restrict__ pam, const float* __restrict__ cam,
                       const float* __restrict__ sem, unsigned short* __restrict__ out,
                       long total) {
  long idx = (long)blockIdx.x * 256 + threadIdx.x;
  if (idx >= total) return;
  out[idx] = f2bf((pam[idx] + cam[idx]) * sem[idx]);
}

// ---------------------------------------------------------------------------
// PAM: flash attention over N=4096 pixels, d=8 (padded to K=32), dv=64.
// One wave = 16 query rows; online softmax; logits never hit HBM.
// ---------------------------------------------------------------------------
__global__ __launch_bounds__(256)
void k_pam_flash(const unsigned short* __restrict__ Qp,   // [B*4096][32] bf16
                 const unsigned short* __restrict__ Kp,   // [B*4096][32] bf16
                 const unsigned short* __restrict__ Vt,   // [B][64][4096] bf16
                 const float* __restrict__ hres,          // [B*4096][64] f32
                 const float* __restrict__ gammaPtr,
                 unsigned short* __restrict__ outBf) {    // [B*4096][64] bf16
  __shared__ __align__(16) unsigned short Pl[8][16][32];
  const int lane = threadIdx.x & 31, wave = threadIdx.x >> 5;
  const int hh = lane >> 4, ln = lane & 15;
  const int b  = blockIdx.x >> 5;                 // 32 blocks per batch
  const int qt = (blockIdx.x & 31) * 8 + wave;    // 16-row tile index
  const int qrow0 = qt * 16;

  const unsigned short* qbase = Qp + ((size_t)b * 4096 + qrow0) * 32;
  u32x4 qa0 = *(const u32x4*)(qbase + (size_t)ln * 32 + hh * 16);
  u32x4 qa1 = *(const u32x4*)(qbase + (size_t)ln * 32 + hh * 16 + 8);
  v16bf aq = frag16(qa0, qa1);

  v8f O0 = vzero8(), O1 = vzero8(), O2 = vzero8(), O3 = vzero8();
  float mrow[8], lrow[8];
  for (int v = 0; v < 8; ++v) { mrow[v] = -1e30f; lrow[v] = 0.0f; }

  const unsigned short* kbase = Kp + (size_t)b * 4096 * 32;
  const unsigned short* vtb   = Vt + (size_t)b * 64 * 4096;
  const v8f zc = vzero8();

  for (int j = 0; j < 4096; j += 32) {
    v16bf bk0 = ldfrag32(kbase + (size_t)(j + ln) * 32 + hh * 16);
    v16bf bk1 = ldfrag32(kbase + (size_t)(j + 16 + ln) * 32 + hh * 16);
    if (j + 64 < 4096)
      __builtin_prefetch(kbase + (size_t)(j + 64 + ln) * 32 + hh * 16, 0, 0);
    v8f s0 = wmma_bf16(aq, bk0, zc);
    v8f s1 = wmma_bf16(aq, bk1, zc);

    float corr[8], p0[8], p1[8];
    for (int v = 0; v < 8; ++v) {
      float xm = fmaxf(s0[v], s1[v]);
      xm = fmaxf(xm, __shfl_xor(xm, 1));
      xm = fmaxf(xm, __shfl_xor(xm, 2));
      xm = fmaxf(xm, __shfl_xor(xm, 4));
      xm = fmaxf(xm, __shfl_xor(xm, 8));   // stays in 16-lane half
      float mnew = fmaxf(mrow[v], xm);
      float c = __expf(mrow[v] - mnew);
      mrow[v] = mnew;
      float e0 = __expf(s0[v] - mnew);
      float e1 = __expf(s1[v] - mnew);
      float rs = e0 + e1;
      rs += __shfl_xor(rs, 1);
      rs += __shfl_xor(rs, 2);
      rs += __shfl_xor(rs, 4);
      rs += __shfl_xor(rs, 8);
      lrow[v] = lrow[v] * c + rs;
      corr[v] = c;
      p0[v] = e0; p1[v] = e1;
    }
    for (int v = 0; v < 8; ++v) {
      O0[v] *= corr[v]; O1[v] *= corr[v]; O2[v] *= corr[v]; O3[v] *= corr[v];
    }
    // C-layout P -> LDS -> A-layout fragment (per-wave private region)
    for (int v = 0; v < 8; ++v) {
      Pl[wave][v + 8 * hh][ln]      = f2bf(p0[v]);
      Pl[wave][v + 8 * hh][16 + ln] = f2bf(p1[v]);
    }
    u32x4 pa0 = *(const u32x4*)&Pl[wave][ln][hh * 16];
    u32x4 pa1 = *(const u32x4*)&Pl[wave][ln][hh * 16 + 8];
    v16bf ap = frag16(pa0, pa1);

    O0 = wmma_bf16(ap, ldfrag32(vtb + (size_t)(0  + ln) * 4096 + j + hh * 16), O0);
    O1 = wmma_bf16(ap, ldfrag32(vtb + (size_t)(16 + ln) * 4096 + j + hh * 16), O1);
    O2 = wmma_bf16(ap, ldfrag32(vtb + (size_t)(32 + ln) * 4096 + j + hh * 16), O2);
    O3 = wmma_bf16(ap, ldfrag32(vtb + (size_t)(48 + ln) * 4096 + j + hh * 16), O3);
  }

  const float g = gammaPtr[0];
  for (int v = 0; v < 8; ++v) {
    int row = qrow0 + v + 8 * hh;
    float inv = 1.0f / lrow[v];
    size_t base = ((size_t)b * 4096 + row) * 64;
    outBf[base + 0  + ln] = f2bf(g * (O0[v] * inv) + hres[base + 0  + ln]);
    outBf[base + 16 + ln] = f2bf(g * (O1[v] * inv) + hres[base + 16 + ln]);
    outBf[base + 32 + ln] = f2bf(g * (O2[v] * inv) + hres[base + 32 + ln]);
    outBf[base + 48 + ln] = f2bf(g * (O3[v] * inv) + hres[base + 48 + ln]);
  }
}

// ---------------------------------------------------------------------------
// CAM: Gram (H^T H, 64x64, K=4096) via WMMA from channel-major bf16.
// ---------------------------------------------------------------------------
__global__ __launch_bounds__(256)
void k_cam_gram(const unsigned short* __restrict__ Ht, float* __restrict__ gram) {
  const int lane = threadIdx.x & 31, wave = threadIdx.x >> 5;
  const int hh = lane >> 4, ln = lane & 15;
  const int b = blockIdx.x;
  const unsigned short* base = Ht + (size_t)b * 64 * 4096;
  for (int t = wave; t < 16; t += 8) {
    int ct = t >> 2, dt = t & 3;
    v8f acc = vzero8();
    const unsigned short* ar = base + (size_t)(ct * 16 + ln) * 4096;
    const unsigned short* br = base + (size_t)(dt * 16 + ln) * 4096;
    for (int kb = 0; kb < 4096; kb += 32) {
      u32x4 a0 = *(const u32x4*)(ar + kb + hh * 16);
      u32x4 a1 = *(const u32x4*)(ar + kb + hh * 16 + 8);
      v16bf afrag = frag16(a0, a1);
      v16bf bfrag = ldfrag32(br + kb + hh * 16);
      acc = wmma_bf16(afrag, bfrag, acc);
    }
    for (int v = 0; v < 8; ++v)
      gram[((size_t)b * 64 + ct * 16 + v + 8 * hh) * 64 + dt * 16 + ln] = acc[v];
  }
}

__global__ void k_cam_softmax(const float* __restrict__ gram,
                              unsigned short* __restrict__ att) {
  __shared__ float red[64];
  const int c = blockIdx.x & 63, b = blockIdx.x >> 6;
  const int d = threadIdx.x;
  float e = gram[((size_t)b * 64 + c) * 64 + d];
  red[d] = e; __syncthreads();
  float M = -1e30f;
  for (int i = 0; i < 64; ++i) M = fmaxf(M, red[i]);
  __syncthreads();
  float val = M - e;                       // max - e
  red[d] = val; __syncthreads();
  float m2 = -1e30f;
  for (int i = 0; i < 64; ++i) m2 = fmaxf(m2, red[i]);
  __syncthreads();
  float p = __expf(val - m2);
  red[d] = p; __syncthreads();
  float s = 0.0f;
  for (int i = 0; i < 64; ++i) s += red[i];
  att[((size_t)b * 64 + c) * 64 + d] = f2bf(p / s);
}

// o[i][c] = gamma * sum_d H[i][d]*att[c][d] + h[i][c]   (M=4096, N=64, K=64)
__global__ __launch_bounds__(256)
void k_cam_apply(const unsigned short* __restrict__ hbf,  // [B*4096][64] bf16
                 const unsigned short* __restrict__ att,  // [B][64][64] bf16
                 const float* __restrict__ hres,          // [B*4096][64] f32
                 const float* __restrict__ gammaPtr,
                 unsigned short* __restrict__ outBf) {
  const int lane = threadIdx.x & 31, wave = threadIdx.x >> 5;
  const int hh = lane >> 4, ln = lane & 15;
  int tile = blockIdx.x * 8 + wave;
  if (tile >= 4 * 256 * 4) return;
  int nt = tile & 3;
  int mt = tile >> 2;
  int b = mt >> 8;
  int mloc = (mt & 255) * 16;
  const unsigned short* hb = hbf + (size_t)b * 4096 * 64;
  const unsigned short* ab = att + (size_t)b * 64 * 64;
  v8f acc = vzero8();
  for (int kb = 0; kb < 64; kb += 32) {
    u32x4 a0 = *(const u32x4*)(hb + (size_t)(mloc + ln) * 64 + kb + hh * 16);
    u32x4 a1 = *(const u32x4*)(hb + (size_t)(mloc + ln) * 64 + kb + hh * 16 + 8);
    v16bf afrag = frag16(a0, a1);
    v16bf bfrag = ldfrag32(ab + (size_t)(nt * 16 + ln) * 64 + kb + hh * 16);
    acc = wmma_bf16(afrag, bfrag, acc);
  }
  const float g = gammaPtr[0];
  for (int v = 0; v < 8; ++v) {
    size_t row = (size_t)b * 4096 + mloc + v + 8 * hh;
    int ch = nt * 16 + ln;
    outBf[row * 64 + ch] = f2bf(g * acc[v] + hres[row * 64 + ch]);
  }
}

// ---------------------------------------------------------------------------
// Host-side orchestration
// ---------------------------------------------------------------------------
static inline int ilog2i(int x) { int s = 0; while ((1 << s) < x) ++s; return s; }

extern "C" void kernel_launch(void* const* d_in, const int* in_sizes, int n_in,
                              void* d_out, int out_size, void* d_ws, size_t ws_size,
                              hipStream_t stream) {
  (void)in_sizes; (void)n_in; (void)out_size; (void)ws_size;
#define IN(i) ((const float*)d_in[i])

  char* cur = (char*)d_ws;
  auto alloc = [&](size_t bytes) -> void* {
    void* p = (void*)cur;
    cur += (bytes + 255) & ~(size_t)255;
    return p;
  };

  const long NPIX = 4L * 64 * 64;      // 16384 pixels total at 64x64
  const long NP4  = 4L * 32 * 32;      // 4096 at 32x32
  const long NP16 = 4L * 16 * 16;      // 1024 at 16x16

  // Activations / intermediates
  unsigned short* cmb_bf  = (unsigned short*)alloc(NPIX * 128 * 2);
  unsigned short* e1a_bf  = (unsigned short*)alloc(NPIX * 256 * 2);
  float*          e1b_f   = (float*)        alloc(NPIX * 256 * 4);
  unsigned short* p1_bf   = (unsigned short*)alloc(NP4  * 256 * 2);
  unsigned short* e2a_bf  = (unsigned short*)alloc(NP4  * 512 * 2);
  float*          e2b_f   = (float*)        alloc(NP4  * 512 * 4);
  float*          p2_f    = (float*)        alloc(NP16 * 512 * 4);
  unsigned short* up2_bf  = (unsigned short*)alloc(NP4  * 512 * 2);
  unsigned short* d2a_bf  = (unsigned short*)alloc(NP4  * 256 * 2);
  float*          d2b_f   = (float*)        alloc(NP4  * 256 * 4);
  unsigned short* up1_bf  = (unsigned short*)alloc(NPIX * 256 * 2);
  unsigned short* d1a_bf  = (unsigned short*)alloc(NPIX * 128 * 2);
  unsigned short* sem_bf  = (unsigned short*)alloc(NPIX * 128 * 2);
  float*          csem_f  = (float*)        alloc(NPIX * 64 * 4);
  float*          ph_f    = (float*)        alloc(NPIX * 64 * 4);
  unsigned short* ph_bf   = (unsigned short*)alloc(NPIX * 64 * 2);
  float*          q_f     = (float*)        alloc(NPIX * 8 * 4);
  float*          kk_f    = (float*)        alloc(NPIX * 8 * 4);
  float*          v_f     = (float*)        alloc(NPIX * 64 * 4);
  unsigned short* Qp      = (unsigned short*)alloc(NPIX * 32 * 2);
  unsigned short* Kp      = (unsigned short*)alloc(NPIX * 32 * 2);
  unsigned short* Vt      = (unsigned short*)alloc(NPIX * 64 * 2);
  unsigned short* pat_bf  = (unsigned short*)alloc(NPIX * 64 * 2);
  float*          pout_f  = (float*)        alloc(NPIX * 64 * 4);
  float*          ch_f    = (float*)        alloc(NPIX * 64 * 4);
  unsigned short* ch_bf   = (unsigned short*)alloc(NPIX * 64 * 2);
  unsigned short* Htc     = (unsigned short*)alloc(NPIX * 64 * 2);
  float*          gram_f  = (float*)        alloc(4L * 64 * 64 * 4);
  unsigned short* att_bf  = (unsigned short*)alloc(4L * 64 * 64 * 2);
  unsigned short* cat_bf  = (unsigned short*)alloc(NPIX * 64 * 2);
  float*          cout_f  = (float*)        alloc(NPIX * 64 * 4);
  unsigned short* fuse_bf = (unsigned short*)alloc(NPIX * 64 * 2);

  auto packw = [&](int widx, int Cin, int Cout, int ks) -> unsigned short* {
    int Nt = (Cout + 15) / 16;
    long K = (long)ks * ks * Cin;
    long elems = K * Nt * 16;
    unsigned short* wp = (unsigned short*)alloc((size_t)elems * 2);
    k_pack_w<<<(int)((elems + 255) / 256), 256, 0, stream>>>(
        IN(widx), wp, Cin, Cout, ks, Nt * 16, elems);
    return wp;
  };

  auto conv = [&](const unsigned short* act, const unsigned short* wpk,
                  const float* bias, const float* bnS, const float* bnB,
                  const float* slope, float* outF, unsigned short* outBf,
                  int B, int H, int W, int Cin, int Cout, int ks,
                  int actMode, int storeNCHW) {
    ConvCfg c;
    c.H = H; c.W = W; c.HW = H * W; c.M = B * H * W;
    c.hwShift = ilog2i(H * W); c.wShift = ilog2i(W); c.cinShift = ilog2i(Cin);
    c.Cin = Cin; c.Cout = Cout; c.K = ks * ks * Cin; c.pad = ks / 2;
    c.Ntiles = (Cout + 15) / 16; c.act = actMode; c.storeNCHW = storeNCHW;
    int Mtiles = c.M >> 4;                  // always a multiple of 8 here
    int blocks = (Mtiles / 8) * c.Ntiles;
    k_conv_wmma<<<blocks, 256, 0, stream>>>(
        act, wpk, bias, bnS, bnB, slope, outF, outBf, c);
  };

#define EW(kern, total, ...) \
  kern<<<(int)(((total) + 255) / 256), 256, 0, stream>>>(__VA_ARGS__)

  // --- pack all weights (deterministic every call) ---
  unsigned short* w_e1c1 = packw(2, 128, 256, 3);
  unsigned short* w_e1c2 = packw(6, 256, 256, 3);
  unsigned short* w_e2c1 = packw(10, 256, 512, 3);
  unsigned short* w_e2c2 = packw(14, 512, 512, 3);
  unsigned short* w_d2c1 = packw(18, 512, 256, 3);
  unsigned short* w_d2c2 = packw(22, 256, 256, 3);
  unsigned short* w_d1c1 = packw(26, 256, 128, 3);
  unsigned short* w_d1c2 = packw(30, 128, 128, 3);
  unsigned short* w_pc0  = packw(34, 128, 64, 3);
  unsigned short* w_pc1  = packw(39, 64, 64, 3);
  unsigned short* w_q    = packw(45, 64, 8, 1);
  unsigned short* w_k    = packw(47, 64, 8, 1);
  unsigned short* w_v    = packw(49, 64, 64, 1);
  unsigned short* w_cc0  = packw(51, 128, 64, 3);
  unsigned short* w_cc1  = packw(56, 64, 64, 3);
  unsigned short* w_cv   = packw(62, 64, 64, 1);
  unsigned short* w_cs   = packw(64, 128, 64, 3);

  // --- combined = concat(x, fused) -> bf16 NHWC [4,64,64,128] ---
  EW(k_concat, NPIX * 128, IN(0), IN(1), cmb_bf, NPIX * 128);

  // --- semantic U-Net ---
  conv(cmb_bf, w_e1c1, IN(3), IN(4), IN(5), nullptr, nullptr, e1a_bf, 4, 64, 64, 128, 256, 3, 1, 0);
  conv(e1a_bf, w_e1c2, IN(7), IN(8), IN(9), nullptr, e1b_f, nullptr, 4, 64, 64, 256, 256, 3, 1, 0);
  EW(k_maxpool2, NP4 * 256, e1b_f, nullptr, p1_bf, 64, 64, 256, NP4 * 256);
  conv(p1_bf, w_e2c1, IN(11), IN(12), IN(13), nullptr, nullptr, e2a_bf, 4, 32, 32, 256, 512, 3, 1, 0);
  conv(e2a_bf, w_e2c2, IN(15), IN(16), IN(17), nullptr, e2b_f, nullptr, 4, 32, 32, 512, 512, 3, 1, 0);
  EW(k_maxpool2, NP16 * 512, e2b_f, p2_f, nullptr, 32, 32, 512, NP16 * 512);
  EW(k_upsample2, NP4 * 512, p2_f, up2_bf, 16, 16, 512, NP4 * 512);
  conv(up2_bf, w_d2c1, IN(19), IN(20), IN(21), nullptr, nullptr, d2a_bf, 4, 32, 32, 512, 256, 3, 1, 0);
  conv(d2a_bf, w_d2c2, IN(23), IN(24), IN(25), nullptr, d2b_f, nullptr, 4, 32, 32, 256, 256, 3, 1, 0);
  EW(k_upsample2, NPIX * 256, d2b_f, up1_bf, 32, 32, 256, NPIX * 256);
  conv(up1_bf, w_d1c1, IN(27), IN(28), IN(29), nullptr, nullptr, d1a_bf, 4, 64, 64, 256, 128, 3, 1, 0);
  conv(d1a_bf, w_d1c2, IN(31), IN(32), IN(33), nullptr, nullptr, sem_bf, 4, 64, 64, 128, 128, 3, 1, 0);
  conv(sem_bf, w_cs, IN(65), nullptr, nullptr, nullptr, csem_f, nullptr, 4, 64, 64, 128, 64, 3, 0, 0);

  // --- PAM branch ---
  conv(cmb_bf, w_pc0, IN(35), IN(36), IN(37), IN(38), ph_f, ph_bf, 4, 64, 64, 128, 64, 3, 2, 0);
  conv(ph_bf, w_q, IN(46), nullptr, nullptr, nullptr, q_f, nullptr, 4, 64, 64, 64, 8, 1, 0, 0);
  conv(ph_bf, w_k, IN(48), nullptr, nullptr, nullptr, kk_f, nullptr, 4, 64, 64, 64, 8, 1, 0, 0);
  conv(ph_bf, w_v, IN(50), nullptr, nullptr, nullptr, v_f, nullptr, 4, 64, 64, 64, 64, 1, 0, 0);
  EW(k_pad_qk, NPIX * 32, q_f, Qp, NPIX * 32);
  EW(k_pad_qk, NPIX * 32, kk_f, Kp, NPIX * 32);
  EW(k_transpose_cn, NPIX * 64, v_f, Vt, NPIX * 64);
  k_pam_flash<<<128, 256, 0, stream>>>(Qp, Kp, Vt, ph_f, IN(44), pat_bf);
  conv(pat_bf, w_pc1, IN(40), IN(41), IN(42), IN(43), pout_f, nullptr, 4, 64, 64, 64, 64, 3, 2, 0);

  // --- CAM branch ---
  conv(cmb_bf, w_cc0, IN(52), IN(53), IN(54), IN(55), ch_f, ch_bf, 4, 64, 64, 128, 64, 3, 2, 0);
  EW(k_transpose_cn, NPIX * 64, ch_f, Htc, NPIX * 64);
  k_cam_gram<<<4, 256, 0, stream>>>(Htc, gram_f);
  k_cam_softmax<<<256, 64, 0, stream>>>(gram_f, att_bf);
  k_cam_apply<<<512, 256, 0, stream>>>(ch_bf, att_bf, ch_f, IN(61), cat_bf);
  conv(cat_bf, w_cc1, IN(57), IN(58), IN(59), IN(60), cout_f, nullptr, 4, 64, 64, 64, 64, 3, 2, 0);

  // --- fuse + final 1x1 conv (writes NCHW f32 directly to d_out) ---
  EW(k_fuse, NPIX * 64, pout_f, cout_f, csem_f, fuse_bf, NPIX * 64);
  conv(fuse_bf, w_cv, IN(63), nullptr, nullptr, nullptr, (float*)d_out, nullptr,
       4, 64, 64, 64, 64, 1, 0, 1);

#undef EW
#undef IN
}
